// EdgeBlock_90529320665781
// MI455X (gfx1250) — matrix operand order repaired and checked by reading, not compiled, not run on hardware
//
#include <hip/hip_runtime.h>
#include <stdint.h>

typedef __attribute__((ext_vector_type(16))) _Float16    v16h;
typedef __attribute__((ext_vector_type(8)))  float        v8f;
typedef __attribute__((ext_vector_type(8)))  unsigned int v8u;

#define E_TILE 128          // edges per block
#define NWAVES 4            // 128 threads, wave32
// per-wave M = 32 (two 16-row A fragments) => B fragments reused 2x

// packed-weight layout offsets in d_ws (in dwords).  Each B fragment
// (K=32 x N=16 f16) = 256 dwords, stored [lane][vgpr0..7] so a lane
// fetches its 8 dwords with one 32-byte load.
#define OFF_WD   0        // W_dist : 16 ktiles x 8 ntiles
#define OFF_WE   32768    // W_edge :  4 ktiles x 8 ntiles
#define OFF_WG1  40960    // W_g1   :  4 ktiles x 2 ntiles
#define OFF_WG2  43008    // W_g2   :  1 ktile  x 40 ntiles
#define OFF_CG   53248    // [cg_mid | cg_all] : 8 ktiles x 2 ntiles (256 x 25 pad 32)
#define PACK_DWORDS 57344

// dynamic LDS partition (bytes)
#define LDS_XEH   0                       // 128*128 f16 = 32768
#define LDS_GC    32768                   // 128*128 f16 = 32768 (combo, then gate0)
#define LDS_TP    65536                   // 128*32  f32 = 16384
#define LDS_H     81920                   // 128*32  f16 = 8192
#define LDS_TOTAL 90112

__device__ __forceinline__ float sigf(float x){ return 1.0f/(1.0f+__expf(-x)); }

__device__ __forceinline__ unsigned int pk2(float a, float b){
  _Float16 ha=(_Float16)a, hb=(_Float16)b;
  unsigned short ua, ub;
  __builtin_memcpy(&ua,&ha,2); __builtin_memcpy(&ub,&hb,2);
  return (unsigned int)ua | ((unsigned int)ub<<16);
}

__device__ __forceinline__ v16h load_frag(const unsigned int* p, int frag, int lane){
  return *reinterpret_cast<const v16h*>(p + frag*256 + lane*8);
}

__device__ __forceinline__ v8f wmma16(const v16h& a, const v16h& b, const v8f& c){
  return __builtin_amdgcn_wmma_f32_16x16x32_f16(false, a, false, b, (short)0, c, false, false);
}

// ---------------------------------------------------------------------------
// Prep: pack all weights into f16 WMMA B-fragment layout (run each launch).
// B (32x16) striping: lanes 0-15 -> N=lane, K=0..7,16..23 (2/vgpr); lanes 16-31 ->
// N=lane-16, K=8..15,24..31.
// ---------------------------------------------------------------------------
__global__ void pack_weights_kernel(const float* __restrict__ Wd,
                                    const float* __restrict__ We,
                                    const float* __restrict__ Wg1,
                                    const float* __restrict__ Wg2,
                                    const float* __restrict__ cm,
                                    const float* __restrict__ ca,
                                    unsigned int* __restrict__ ws){
  int t = blockIdx.x*blockDim.x + threadIdx.x;
  if (t >= PACK_DWORDS) return;
  int d, ntiles, ldN, region; const float* src = nullptr;
  if      (t < OFF_WE ){ d=t;          ntiles=8;  ldN=128; src=Wd;  region=0; }
  else if (t < OFF_WG1){ d=t-OFF_WE;   ntiles=8;  ldN=128; src=We;  region=0; }
  else if (t < OFF_WG2){ d=t-OFF_WG1;  ntiles=2;  ldN=32;  src=Wg1; region=0; }
  else if (t < OFF_CG ){ d=t-OFF_WG2;  ntiles=40; ldN=640; src=Wg2; region=0; }
  else                 { d=t-OFF_CG;   ntiles=2;  ldN=0;             region=1; }
  int frag = d>>8, wi = d&255, ln = wi>>3, v = wi&7;
  int kt = frag/ntiles, nt = frag - kt*ntiles;
  int n  = (ln&15) + nt*16;
  int k0 = kt*32 + ((v&4)<<2) + ((ln>>4)<<3) + 2*(v&3);
  float w0, w1;
  if (region==0){
    w0 = src[(size_t)k0*ldN + n];
    w1 = src[(size_t)(k0+1)*ldN + n];
  } else {
    auto cg = [&](int kk)->float{       // kk = i*16+j, flat over (SB,SB)
      if (n < 9)  return cm[kk*9  + n];
      if (n < 25) return ca[kk*16 + (n-9)];
      return 0.0f;
    };
    w0 = cg(k0); w1 = cg(k0+1);
  }
  ws[t] = pk2(w0, w1);
}

// ---------------------------------------------------------------------------
// Main fused kernel: 128 threads (4 waves) per block, 128 edges per block,
// each wave owns 32 edge rows (two M=16 fragments) => every B fragment loaded
// once per wave feeds two WMMAs (halves the L2->WGP B-traffic per FLOP).
// ---------------------------------------------------------------------------
__global__ __launch_bounds__(128) void edge_block_kernel(
    const float* __restrict__ edge_distance,
    const int*   __restrict__ srcel, const int* __restrict__ tgtel,
    const float* __restrict__ x_sph, const float* __restrict__ edge_sh,
    const float* __restrict__ b_dist,
    const float* __restrict__ src_emb, const float* __restrict__ tgt_emb,
    const float* __restrict__ b_edge,  const float* __restrict__ b_g1,
    const float* __restrict__ b_g2,    const float* __restrict__ W_path,
    const unsigned int* __restrict__ ws,
    float* __restrict__ out, int E)
{
  extern __shared__ __align__(16) char smem[];
  _Float16* sh_xeh = (_Float16*)(smem + LDS_XEH);   // x_edge staging (f16)
  _Float16* sh_gc  = (_Float16*)(smem + LDS_GC);    // combo, then gate0 (f16)
  float*    sh_tp  = (float*)   (smem + LDS_TP);    // tp_mid|tp_all|S
  _Float16* sh_h   = (_Float16*)(smem + LDS_H);     // h (f16)

  const int tid  = threadIdx.x;
  const int wave = tid>>5, lane = tid&31;
  const int wb   = wave*32;                 // wave's edge-row base in tile
  const int base = blockIdx.x*E_TILE;
  const int bo   = (lane>>4)*8;             // K half-lane offset (A/B layout)
  const int mrow = (lane>>4)*8;             // C-layout M offset
  const int ncol = lane & 15;               // C-layout N-within-tile

  // ---- stage combo = b_dist + src_emb[src] + tgt_emb[tgt]  (f16) ----
  for (int i = tid; i < E_TILE*32; i += 128){
    int e = i>>5, c4 = i&31;
    int eg = min(base+e, E-1);
    int si = srcel[eg], ti = tgtel[eg];
    float4 bd = *(const float4*)(b_dist + c4*4);
    float4 se = *(const float4*)(src_emb + (size_t)si*128 + c4*4);
    float4 te = *(const float4*)(tgt_emb + (size_t)ti*128 + c4*4);
    unsigned int* p = (unsigned int*)(sh_gc + e*128 + c4*4);
    p[0] = pk2(bd.x+se.x+te.x, bd.y+se.y+te.y);
    p[1] = pk2(bd.z+se.z+te.z, bd.w+se.w+te.w);
  }
  __syncthreads();

  const float delta = 6.0f/511.0f;
  const float coeff = -0.5f/(delta*delta);
  const float drow0 = edge_distance[min(base + wb +      ncol, E-1)];
  const float drow1 = edge_distance[min(base + wb + 16 + ncol, E-1)];

  // ================= GEMM1: g(Ex512) @ W_dist(512x128) ==================
  // A synthesized in registers (Gaussian radial basis), 16 K-steps,
  // two M=16 fragments per wave sharing each B fragment.
  v8f acc[2][8];
  #pragma unroll
  for (int mt=0; mt<2; ++mt)
    #pragma unroll
    for (int nt=0; nt<8; ++nt){ v8f z = {}; acc[mt][nt] = z; }
  for (int kt=0; kt<16; ++kt){
    v16h a0, a1;
    #pragma unroll
    for (int v=0; v<8; ++v){
      int   klc = ((v&4)<<2) + 2*(v&3);
      float k0f = (float)(kt*32 + klc + bo);
      float o0  = k0f*delta, o1 = (k0f+1.0f)*delta;
      float d00 = drow0-o0, d01 = drow0-o1;
      float d10 = drow1-o0, d11 = drow1-o1;
      a0[2*v]   = (_Float16)__expf(coeff*d00*d00);
      a0[2*v+1] = (_Float16)__expf(coeff*d01*d01);
      a1[2*v]   = (_Float16)__expf(coeff*d10*d10);
      a1[2*v+1] = (_Float16)__expf(coeff*d11*d11);
    }
    #pragma unroll
    for (int nt=0; nt<8; ++nt){
      v16h b = load_frag(ws+OFF_WD, kt*8+nt, lane);
      acc[0][nt] = wmma16(a0, b, acc[0][nt]);
      acc[1][nt] = wmma16(a1, b, acc[1][nt]);
    }
  }
  // epilogue: + combo, SiLU -> f16 LDS (rows owned by this wave)
  #pragma unroll
  for (int mt=0; mt<2; ++mt){
    #pragma unroll
    for (int nt=0; nt<8; ++nt){
      #pragma unroll
      for (int r=0; r<8; ++r){
        int m = wb + mt*16 + mrow + r, n = ncol + nt*16;
        float v = acc[mt][nt][r] + (float)sh_gc[m*128+n];
        sh_xeh[m*128+n] = (_Float16)(v*sigf(v));
      }
    }
  }
  __syncthreads();   // combo fully consumed; sh_gc reusable for gate

  const unsigned int* xl = (const unsigned int*)sh_xeh;
  const int arow0 = (wb +      ncol)*128;
  const int arow1 = (wb + 16 + ncol)*128;

  // ================= GEMM2: x0(Ex128) @ W_edge(128x128) ================
  #pragma unroll
  for (int mt=0; mt<2; ++mt)
    #pragma unroll
    for (int nt=0; nt<8; ++nt){ v8f z = {}; acc[mt][nt] = z; }
  for (int kt=0; kt<4; ++kt){
    v8u au0, au1;
    #pragma unroll
    for (int v=0; v<8; ++v){
      int kpb = ((v&4)<<2) + 2*(v&3) + bo;
      au0[v] = xl[(arow0 + kt*32 + kpb)>>1];
      au1[v] = xl[(arow1 + kt*32 + kpb)>>1];
    }
    v16h a0 = __builtin_bit_cast(v16h, au0);
    v16h a1 = __builtin_bit_cast(v16h, au1);
    #pragma unroll
    for (int nt=0; nt<8; ++nt){
      v16h b = load_frag(ws+OFF_WE, kt*8+nt, lane);
      acc[0][nt] = wmma16(a0, b, acc[0][nt]);
      acc[1][nt] = wmma16(a1, b, acc[1][nt]);
    }
  }
  #pragma unroll
  for (int nt=0; nt<8; ++nt){
    float bn = b_edge[ncol + nt*16];
    #pragma unroll
    for (int mt=0; mt<2; ++mt){
      #pragma unroll
      for (int r=0; r<8; ++r){
        int m = wb + mt*16 + mrow + r, n = ncol + nt*16;
        float v = acc[mt][nt][r] + bn;
        sh_xeh[m*128+n] = (_Float16)(v*sigf(v));   // x_edge (kept for output)
      }
    }
  }

  // ================= GEMM3: x_edge(Ex128) @ W_g1(128x32), ReLU =========
  v8f h0[2], h1[2];
  { v8f z = {}; h0[0]=z; h0[1]=z; h1[0]=z; h1[1]=z; }
  for (int kt=0; kt<4; ++kt){
    v8u au0, au1;
    #pragma unroll
    for (int v=0; v<8; ++v){
      int kpb = ((v&4)<<2) + 2*(v&3) + bo;
      au0[v] = xl[(arow0 + kt*32 + kpb)>>1];
      au1[v] = xl[(arow1 + kt*32 + kpb)>>1];
    }
    v16h a0 = __builtin_bit_cast(v16h, au0);
    v16h a1 = __builtin_bit_cast(v16h, au1);
    v16h bb0 = load_frag(ws+OFF_WG1, kt*2+0, lane);
    v16h bb1 = load_frag(ws+OFF_WG1, kt*2+1, lane);
    h0[0] = wmma16(a0, bb0, h0[0]);  h0[1] = wmma16(a1, bb0, h0[1]);
    h1[0] = wmma16(a0, bb1, h1[0]);  h1[1] = wmma16(a1, bb1, h1[1]);
  }
  #pragma unroll
  for (int nt=0; nt<2; ++nt){
    float bn = b_g1[ncol + nt*16];
    #pragma unroll
    for (int mt=0; mt<2; ++mt){
      #pragma unroll
      for (int r=0; r<8; ++r){
        int m = wb + mt*16 + mrow + r, n = ncol + nt*16;
        float v = (nt ? h1[mt][r] : h0[mt][r]) + bn;
        sh_h[m*32+n] = (_Float16)fmaxf(v, 0.0f);
      }
    }
  }

  // ============ GEMM4: h(Ex32) @ W_g2(32x640) + softmax over BR=5 ======
  {
    const unsigned int* hl = (const unsigned int*)sh_h;
    int hrow0 = (wb +      ncol)*32;
    int hrow1 = (wb + 16 + ncol)*32;
    v8u au0, au1;
    #pragma unroll
    for (int v=0; v<8; ++v){
      int kpb = ((v&4)<<2) + 2*(v&3) + bo;
      au0[v] = hl[(hrow0 + kpb)>>1];
      au1[v] = hl[(hrow1 + kpb)>>1];
    }
    v16h ah0 = __builtin_bit_cast(v16h, au0);
    v16h ah1 = __builtin_bit_cast(v16h, au1);
    for (int t=0; t<8; ++t){             // 8 column tiles of C=128
      v8f z[2][5];
      #pragma unroll
      for (int b5=0; b5<5; ++b5){
        v16h b = load_frag(ws+OFF_WG2, b5*8+t, lane);
        v8f zc = {};
        v8f z0 = wmma16(ah0, b, zc);
        v8f z1 = wmma16(ah1, b, zc);
        float bn = b_g2[b5*128 + ncol + t*16];
        #pragma unroll
        for (int r=0; r<8; ++r){ z0[r] += bn; z1[r] += bn; }
        z[0][b5] = z0; z[1][b5] = z1;
      }
      #pragma unroll
      for (int mt=0; mt<2; ++mt){
        #pragma unroll
        for (int r=0; r<8; ++r){
          float mx = z[mt][0][r];
          #pragma unroll
          for (int b5=1; b5<5; ++b5) mx = fmaxf(mx, z[mt][b5][r]);
          float s = 0.0f;
          #pragma unroll
          for (int b5=0; b5<5; ++b5) s += __expf(z[mt][b5][r]-mx);
          float g0 = __expf(z[mt][0][r]-mx)/s;
          int m = wb + mt*16 + mrow + r, n = ncol + t*16;
          sh_gc[m*128+n] = (_Float16)g0;   // gate[:,0,:]
        }
      }
    }
  }

  // ===== TP GEMM: outer(x_sph,edge_sh)(Ex256) @ [cg_mid|cg_all](256x25) =
  float xs0[16], xs1[16], es0[8], es1[8];
  {
    int eg0 = min(base + wb +      ncol, E-1);
    int eg1 = min(base + wb + 16 + ncol, E-1);
    const float4* xp0 = (const float4*)(x_sph + (size_t)eg0*16);
    const float4* xp1 = (const float4*)(x_sph + (size_t)eg1*16);
    #pragma unroll
    for (int q=0; q<4; ++q){
      float4 x = xp0[q];
      xs0[4*q+0]=x.x; xs0[4*q+1]=x.y; xs0[4*q+2]=x.z; xs0[4*q+3]=x.w;
      float4 y = xp1[q];
      xs1[4*q+0]=y.x; xs1[4*q+1]=y.y; xs1[4*q+2]=y.z; xs1[4*q+3]=y.w;
    }
    const float4* ep0 = (const float4*)(edge_sh + (size_t)eg0*16 + bo);
    const float4* ep1 = (const float4*)(edge_sh + (size_t)eg1*16 + bo);
    #pragma unroll
    for (int q=0; q<2; ++q){
      float4 x = ep0[q];
      es0[4*q+0]=x.x; es0[4*q+1]=x.y; es0[4*q+2]=x.z; es0[4*q+3]=x.w;
      float4 y = ep1[q];
      es1[4*q+0]=y.x; es1[4*q+1]=y.y; es1[4*q+2]=y.z; es1[4*q+3]=y.w;
    }
  }
  v8f tac[2][2];
  { v8f z = {}; tac[0][0]=z; tac[0][1]=z; tac[1][0]=z; tac[1][1]=z; }
  #pragma unroll
  for (int kt=0; kt<8; ++kt){
    v16h a0, a1;
    #pragma unroll
    for (int v=0; v<8; ++v){
      int ih = (v>>2)&1, jc = 2*(v&3);
      float x0 = xs0[2*kt + ih], x1 = xs1[2*kt + ih];
      a0[2*v]   = (_Float16)(x0*es0[jc]);
      a0[2*v+1] = (_Float16)(x0*es0[jc+1]);
      a1[2*v]   = (_Float16)(x1*es1[jc]);
      a1[2*v+1] = (_Float16)(x1*es1[jc+1]);
    }
    v16h bb0 = load_frag(ws+OFF_CG, kt*2+0, lane);
    v16h bb1 = load_frag(ws+OFF_CG, kt*2+1, lane);
    tac[0][0] = wmma16(a0, bb0, tac[0][0]);  tac[1][0] = wmma16(a1, bb0, tac[1][0]);
    tac[0][1] = wmma16(a0, bb1, tac[0][1]);  tac[1][1] = wmma16(a1, bb1, tac[1][1]);
  }
  #pragma unroll
  for (int mt=0; mt<2; ++mt){
    #pragma unroll
    for (int r=0; r<8; ++r){
      int m = wb + mt*16 + mrow + r;
      sh_tp[m*32 + ncol]      = tac[mt][0][r];  // cols 0..8 tp_mid, 9..24 tp_all
      sh_tp[m*32 + ncol + 16] = tac[mt][1][r];
    }
  }
  __syncthreads();

  // per-edge scalar S = silu(s_all) + sigmoid(s_all)*sum(tp_all[1:])
  {
    int e = tid;                       // E_TILE == blockDim.x == 128
    float sa = sh_tp[e*32 + 9];
    float sum = 0.0f;
    #pragma unroll
    for (int c=10; c<25; ++c) sum += sh_tp[e*32 + c];
    float sg = sigf(sa);
    sh_tp[e*32 + 25] = sa*sg + sg*sum;
  }
  __syncthreads();

  // ====================== output assembly (E,9,128) ====================
  // one pass per edge: gate/x_edge/sigmoid computed once, 9 rows emitted
  int c0 = lane*4;
  float4 wpv0 = *(const float4*)(W_path + c0);
  float4 wpv1 = *(const float4*)(W_path + 128 + c0);
  float4 wpv2 = *(const float4*)(W_path + 256 + c0);
  float wp0a[4] = {wpv0.x, wpv0.y, wpv0.z, wpv0.w};
  float wp1a[4] = {wpv1.x, wpv1.y, wpv1.z, wpv1.w};
  float wp2a[4] = {wpv2.x, wpv2.y, wpv2.z, wpv2.w};
  for (int e = wave; e < E_TILE; e += NWAVES){
    int eg = base + e;
    if (eg >= E) continue;
    float tp0 = sh_tp[e*32];
    float S   = sh_tp[e*32 + 25];
    float tpk[9];
    #pragma unroll
    for (int k=0; k<9; ++k) tpk[k] = sh_tp[e*32 + k];
    float sgg[4], sil[4], bse[4];
    #pragma unroll
    for (int cc=0; cc<4; ++cc){
      int c = c0 + cc;
      float g0 = (float)sh_gc[e*128 + c];
      float xe = (float)sh_xeh[e*128 + c];
      float smid = tp0*wp0a[cc];
      float sg = sigf(smid);
      sgg[cc] = sg*g0;            // sigmoid(s_mid)*gate0, reused for k=1..8
      sil[cc] = smid*sg*g0;       // silu(s_mid)*gate0   (k=0)
      bse[cc] = xe + S;           // x_edge + sum(tp_all_g)
    }
    float* orow = out + (size_t)eg*9*128 + c0;
    {
      float4 o;
      o.x = sil[0]+bse[0]; o.y = sil[1]+bse[1];
      o.z = sil[2]+bse[2]; o.w = sil[3]+bse[3];
      *(float4*)orow = o;
    }
    #pragma unroll
    for (int k=1; k<9; ++k){
      const float* wk = (k<4) ? wp1a : wp2a;   // L_OF_K: 0,1,1,1,2,2,2,2,2
      float4 o;
      o.x = tpk[k]*wk[0]*sgg[0] + bse[0];
      o.y = tpk[k]*wk[1]*sgg[1] + bse[1];
      o.z = tpk[k]*wk[2]*sgg[2] + bse[2];
      o.w = tpk[k]*wk[3]*sgg[3] + bse[3];
      *(float4*)(orow + (size_t)k*128) = o;
    }
  }
}

extern "C" void kernel_launch(void* const* d_in, const int* in_sizes, int n_in,
                              void* d_out, int out_size, void* d_ws, size_t ws_size,
                              hipStream_t stream) {
  const float* edge_distance = (const float*)d_in[0];
  const int*   srcel         = (const int*)  d_in[1];
  const int*   tgtel         = (const int*)  d_in[2];
  const float* x_sph         = (const float*)d_in[3];
  const float* edge_sh       = (const float*)d_in[4];
  const float* W_dist        = (const float*)d_in[5];
  const float* b_dist        = (const float*)d_in[6];
  const float* src_emb       = (const float*)d_in[7];
  const float* tgt_emb       = (const float*)d_in[8];
  const float* W_edge        = (const float*)d_in[9];
  const float* b_edge        = (const float*)d_in[10];
  const float* W_g1          = (const float*)d_in[11];
  const float* b_g1          = (const float*)d_in[12];
  const float* W_g2          = (const float*)d_in[13];
  const float* b_g2          = (const float*)d_in[14];
  const float* cg_mid        = (const float*)d_in[15];
  const float* cg_all        = (const float*)d_in[16];
  const float* W_path        = (const float*)d_in[17];
  float* out = (float*)d_out;
  unsigned int* ws = (unsigned int*)d_ws;
  int E = in_sizes[0];

  pack_weights_kernel<<<(PACK_DWORDS+255)/256, 256, 0, stream>>>(
      W_dist, W_edge, W_g1, W_g2, cg_mid, cg_all, ws);

  int blocks = (E + E_TILE - 1) / E_TILE;
  edge_block_kernel<<<blocks, 128, LDS_TOTAL, stream>>>(
      edge_distance, srcel, tgtel, x_sph, edge_sh,
      b_dist, src_emb, tgt_emb, b_edge, b_g1, b_g2, W_path,
      ws, out, E);
}
